// EdgeEnhancedGraphTransformer2D_17721035063344
// MI455X (gfx1250) — matrix operand order repaired and checked by reading, not compiled, not run on hardware
//
#include <hip/hip_runtime.h>
#include <math.h>

typedef __attribute__((ext_vector_type(16))) __bf16 v16bf;
typedef __attribute__((ext_vector_type(8)))  float  v8f;

#define HD 128
#define NH 8
#define DH 16
#define NL 6
#define FFD 512

// ---------- helpers ----------
__device__ __forceinline__ unsigned f2ord(float f) {
    unsigned u = __float_as_uint(f);
    return (u & 0x80000000u) ? ~u : (u | 0x80000000u);
}
__device__ __forceinline__ float ord2f(unsigned u) {
    return (u & 0x80000000u) ? __uint_as_float(u & 0x7fffffffu) : __uint_as_float(~u);
}
#define NEG_INF_ORD 0x007FFFFFu   // f2ord(-inf)

// ---------- weight prep: Wt[N][Kdst] = bf16(W[k][n]), zero-pad k>=Ksrc ----------
__global__ __launch_bounds__(256)
void transpose_cvt(const float* __restrict__ W, __bf16* __restrict__ Wt,
                   int Ksrc, int Kdst, int Nout)
{
    int t = blockIdx.x * 256 + threadIdx.x;
    if (t >= Nout * Kdst) return;
    int n = t / Kdst, k = t % Kdst;
    Wt[t] = (k < Ksrc) ? (__bf16)W[(long)k * Nout + n] : (__bf16)0.f;
}

// ---------- pad node features [N,48] -> [N,64] ----------
__global__ __launch_bounds__(256)
void pad_rows(const float* __restrict__ src, float* __restrict__ dst,
              int n, int Ksrc, int Kdst)
{
    int t = blockIdx.x * 256 + threadIdx.x;
    if (t >= n * Kdst) return;
    int row = t / Kdst, k = t % Kdst;
    dst[t] = (k < Ksrc) ? src[(long)row * Ksrc + k] : 0.f;
}

// ---------- bf16 WMMA GEMM, no guards: M%16==0, K%32==0, Nout%128==0-per-block ----------
// A [M,K] f32 row-major; Wt [Nout][K] bf16 (pre-transposed); wave tile 16x32 (2 WMMAs).
// Block = 4 waves -> 16 x 128 tile. grid = (Nout/128, M/16).
__global__ __launch_bounds__(128)
void gemm_bf16_wmma(const float* __restrict__ A, const __bf16* __restrict__ Wt,
                    const float* __restrict__ bias, const float* __restrict__ res,
                    float* __restrict__ C, int M, int K, int Nout, int relu)
{
    const int lane   = threadIdx.x & 31;
    const int wave   = threadIdx.x >> 5;
    const int half   = lane >> 4;            // 0/1
    const int l16    = lane & 15;
    const int tile_m = blockIdx.y * 16;
    const int tn0    = blockIdx.x * 128 + wave * 32;    // first of two 16-col subtiles
    const int tn1    = tn0 + 16;

    const int  arow  = tile_m + l16;
    const float* Ap  = A + (long)arow * K;
    const __bf16* B0 = Wt + (long)(tn0 + l16) * K + half * 16;
    const __bf16* B1 = Wt + (long)(tn1 + l16) * K + half * 16;

    v8f acc0 = {0.f,0.f,0.f,0.f,0.f,0.f,0.f,0.f};
    v8f acc1 = {0.f,0.f,0.f,0.f,0.f,0.f,0.f,0.f};

    for (int k0 = 0; k0 < K; k0 += 32) {
        // A frag: lanes hold rows; k0+16h .. +15 contiguous
        const float4* ap0 = (const float4*)(Ap + k0 + half * 16);
        const float4* ap1 = (const float4*)(Ap + k0 + half * 16 + 8);
        float4 f0 = ap0[0], f1 = ap0[1];
        float4 f2 = ap1[0], f3 = ap1[1];
        v16bf a;
        a[0] = (__bf16)f0.x;  a[1] = (__bf16)f0.y;  a[2]  = (__bf16)f0.z;  a[3]  = (__bf16)f0.w;
        a[4] = (__bf16)f1.x;  a[5] = (__bf16)f1.y;  a[6]  = (__bf16)f1.z;  a[7]  = (__bf16)f1.w;
        a[8] = (__bf16)f2.x;  a[9] = (__bf16)f2.y;  a[10] = (__bf16)f2.z;  a[11] = (__bf16)f2.w;
        a[12]= (__bf16)f3.x;  a[13]= (__bf16)f3.y;  a[14] = (__bf16)f3.z;  a[15] = (__bf16)f3.w;
        // B frags: 32 contiguous bytes per lane (pre-transposed bf16)
        v16bf b0 = *(const v16bf*)(B0 + k0);
        v16bf b1 = *(const v16bf*)(B1 + k0);
        acc0 = __builtin_amdgcn_wmma_f32_16x16x32_bf16(false, a, false, b0, (short)0, acc0, false, false);
        acc1 = __builtin_amdgcn_wmma_f32_16x16x32_bf16(false, a, false, b1, (short)0, acc1, false, false);
    }

    const int m0 = tile_m + half * 8;
    const float bs0 = bias[tn0 + l16];
    const float bs1 = bias[tn1 + l16];
    #pragma unroll
    for (int r = 0; r < 8; ++r) {
        long i0 = (long)(m0 + r) * Nout + tn0 + l16;
        long i1 = (long)(m0 + r) * Nout + tn1 + l16;
        float v0 = acc0[r] + bs0;
        float v1 = acc1[r] + bs1;
        if (relu) { v0 = fmaxf(v0, 0.f); v1 = fmaxf(v1, 0.f); }
        if (res)  { v0 += res[i0];       v1 += res[i1]; }
        C[i0] = v0;
        C[i1] = v1;
    }
}

// ---------- LayerNorm: one wave per node (128 features, 4 per lane) ----------
__global__ __launch_bounds__(256)
void ln_kernel(const float* __restrict__ x, const float* __restrict__ g,
               const float* __restrict__ b, float* __restrict__ y, int n)
{
    int node = blockIdx.x * 8 + (threadIdx.x >> 5);
    if (node >= n) return;
    int lane = threadIdx.x & 31;
    const float4 v = ((const float4*)(x + (long)node * HD))[lane];
    float s = v.x + v.y + v.z + v.w;
    #pragma unroll
    for (int off = 16; off > 0; off >>= 1) s += __shfl_xor(s, off, 32);
    const float mean = s * (1.f / HD);
    float dx = v.x - mean, dy = v.y - mean, dz = v.z - mean, dw = v.w - mean;
    float q = dx*dx + dy*dy + dz*dz + dw*dw;
    #pragma unroll
    for (int off = 16; off > 0; off >>= 1) q += __shfl_xor(q, off, 32);
    const float rstd = rsqrtf(q * (1.f / HD) + 1e-5f);
    const float4 gg = ((const float4*)g)[lane];
    const float4 bb = ((const float4*)b)[lane];
    float4 o;
    o.x = dx * rstd * gg.x + bb.x;
    o.y = dy * rstd * gg.y + bb.y;
    o.z = dz * rstd * gg.z + bb.z;
    o.w = dw * rstd * gg.w + bb.w;
    ((float4*)(y + (long)node * HD))[lane] = o;
}

// ---------- fused edge-bias: combW[l]=edge_W@Web[l] (4x8), combB[l]=edge_b@Web[l]+beb[l] ----------
__global__ void build_comb(const float* __restrict__ edge_W, const float* __restrict__ edge_b,
                           const float* __restrict__ Web, const float* __restrict__ beb,
                           float* __restrict__ combW, float* __restrict__ combB)
{
    int t = threadIdx.x;
    if (t < NL * 4 * NH) {
        int l = t / 32, kk = (t / NH) % 4, hh = t % NH;
        float s = 0.f;
        for (int j = 0; j < HD; ++j)
            s += edge_W[kk * HD + j] * Web[((long)l * HD + j) * NH + hh];
        combW[t] = s;
    } else if (t < NL * 4 * NH + NL * NH) {
        int u = t - NL * 4 * NH, l = u / NH, hh = u % NH;
        float s = beb[l * NH + hh];
        for (int j = 0; j < HD; ++j)
            s += edge_b[j] * Web[((long)l * HD + j) * NH + hh];
        combB[u] = s;
    }
}

// ---------- attention init / score / accumulate / normalize ----------
__global__ __launch_bounds__(256)
void init_attn(unsigned* __restrict__ nmax, float* __restrict__ den,
               float* __restrict__ out_un, int n)
{
    int i = blockIdx.x * 256 + threadIdx.x;
    if (i < n * NH) { nmax[i] = NEG_INF_ORD; den[i] = 0.f; }
    if (i < n * HD) out_un[i] = 0.f;
}

__global__ __launch_bounds__(256)
void edge_score(const float* __restrict__ q, const float* __restrict__ k,
                const float* __restrict__ eg, const int* __restrict__ ei,
                const float* __restrict__ combW, const float* __restrict__ combB,
                float* __restrict__ sbuf, unsigned* __restrict__ nmax,
                int Ecnt, float scale)
{
    int t = blockIdx.x * 256 + threadIdx.x;
    if (t >= Ecnt * NH) return;
    int e = t >> 3, hh = t & 7;
    int r = ei[e], c = ei[Ecnt + e];
    const float4* qp = (const float4*)(q + (long)r * HD + hh * DH);
    const float4* kp = (const float4*)(k + (long)c * HD + hh * DH);
    float d = 0.f;
    #pragma unroll
    for (int i = 0; i < 4; ++i) {
        float4 qv = qp[i], kv = kp[i];
        d += qv.x * kv.x + qv.y * kv.y + qv.z * kv.z + qv.w * kv.w;
    }
    const float4 ev = *(const float4*)(eg + (long)e * 4);
    float bias = combB[hh]
               + ev.x * combW[0 * NH + hh] + ev.y * combW[1 * NH + hh]
               + ev.z * combW[2 * NH + hh] + ev.w * combW[3 * NH + hh];
    float s = d * scale + bias;
    sbuf[t] = s;
    atomicMax(nmax + (long)r * NH + hh, f2ord(s));
}

__global__ __launch_bounds__(256)
void edge_accum(const float* __restrict__ v, const int* __restrict__ ei,
                const float* __restrict__ sbuf, const unsigned* __restrict__ nmax,
                float* __restrict__ den, float* __restrict__ out_un, int Ecnt)
{
    int t = blockIdx.x * 256 + threadIdx.x;
    if (t >= Ecnt * NH) return;
    int e = t >> 3, hh = t & 7;
    int r = ei[e], c = ei[Ecnt + e];
    float mx = ord2f(nmax[(long)r * NH + hh]);
    float ex = __expf(sbuf[t] - mx);
    atomicAdd(den + (long)r * NH + hh, ex);
    const float4* vp = (const float4*)(v + (long)c * HD + hh * DH);
    float* op = out_un + (long)r * HD + hh * DH;
    #pragma unroll
    for (int i = 0; i < 4; ++i) {
        float4 vv = vp[i];
        atomicAdd(op + i * 4 + 0, ex * vv.x);
        atomicAdd(op + i * 4 + 1, ex * vv.y);
        atomicAdd(op + i * 4 + 2, ex * vv.z);
        atomicAdd(op + i * 4 + 3, ex * vv.w);
    }
}

__global__ __launch_bounds__(256)
void attn_norm(float* __restrict__ out_un, const float* __restrict__ den, int n)
{
    int t = blockIdx.x * 256 + threadIdx.x;
    if (t >= n * HD) return;
    int node = t >> 7, hh = (t & 127) >> 4;
    out_un[t] = out_un[t] / (den[(long)node * NH + hh] + 1e-10f);
}

// ---------- pooling ----------
__global__ __launch_bounds__(256)
void fzero(float* __restrict__ p, int n)
{
    int t = blockIdx.x * 256 + threadIdx.x;
    if (t < n) p[t] = 0.f;
}

__global__ __launch_bounds__(256)
void pool_accum(const float* __restrict__ x, const int* __restrict__ batch,
                float* __restrict__ gsum, float* __restrict__ cnt, int n)
{
    int t = blockIdx.x * 256 + threadIdx.x;
    if (t >= n * HD) return;
    int node = t >> 7, f = t & 127;
    int g = batch[node];
    atomicAdd(gsum + (long)g * HD + f, x[t]);
    if (f == 0) atomicAdd(cnt + g, 1.f);
}

__global__ __launch_bounds__(256)
void pool_final(const float* __restrict__ gsum, const float* __restrict__ cnt,
                float* __restrict__ out, int gtot)
{
    int t = blockIdx.x * 256 + threadIdx.x;
    if (t >= gtot * HD) return;
    out[t] = gsum[t] / fmaxf(cnt[t >> 7], 1.f);
}

// ---------- host orchestration ----------
extern "C" void kernel_launch(void* const* d_in, const int* in_sizes, int n_in,
                              void* d_out, int out_size, void* d_ws, size_t ws_size,
                              hipStream_t stream)
{
    const float* h      = (const float*)d_in[0];
    const float* eg     = (const float*)d_in[1];
    const int*   ei     = (const int*)  d_in[2];
    const int*   batch  = (const int*)  d_in[3];
    const float* node_W = (const float*)d_in[4];
    const float* node_b = (const float*)d_in[5];
    const float* edge_W = (const float*)d_in[6];
    const float* edge_b = (const float*)d_in[7];
    const float* Wq = (const float*)d_in[8],  *bq = (const float*)d_in[9];
    const float* Wk = (const float*)d_in[10], *bk = (const float*)d_in[11];
    const float* Wv = (const float*)d_in[12], *bv = (const float*)d_in[13];
    const float* Wo = (const float*)d_in[14], *bo = (const float*)d_in[15];
    const float* Web = (const float*)d_in[16], *beb = (const float*)d_in[17];
    const float* W1 = (const float*)d_in[18], *b1 = (const float*)d_in[19];
    const float* W2 = (const float*)d_in[20], *b2 = (const float*)d_in[21];
    const float* g1 = (const float*)d_in[22], *be1 = (const float*)d_in[23];
    const float* g2 = (const float*)d_in[24], *be2 = (const float*)d_in[25];
    const float* gF = (const float*)d_in[26], *bF  = (const float*)d_in[27];

    const int ND   = in_sizes[4] / HD;                    // 48
    const int NDp  = (ND + 31) & ~31;                     // 64 (pad to K%32==0)
    const int N    = in_sizes[0] / ND;
    const int Ecnt = in_sizes[2] / 2;
    const int G    = out_size / HD - N;

    // workspace carve
    char* w = (char*)d_ws;
    auto carve = [&](size_t bytes) -> void* {
        void* p = (void*)w; w += (bytes + 255) & ~(size_t)255; return p;
    };
    float*    x      = (float*)   carve((size_t)N * HD * 4);
    float*    xn     = (float*)   carve((size_t)N * HD * 4);
    float*    qb     = (float*)   carve((size_t)N * HD * 4);
    float*    kb     = (float*)   carve((size_t)N * HD * 4);
    float*    vb     = (float*)   carve((size_t)N * HD * 4);
    float*    out_un = (float*)   carve((size_t)N * HD * 4);
    float*    ffn_t  = (float*)   carve((size_t)N * FFD * 4);
    float*    hpad   = (float*)   carve((size_t)N * NDp * 4);
    float*    sbuf   = (float*)   carve((size_t)Ecnt * NH * 4);
    unsigned* nmax   = (unsigned*)carve((size_t)N * NH * 4);
    float*    den    = (float*)   carve((size_t)N * NH * 4);
    float*    combW  = (float*)   carve((size_t)NL * 4 * NH * 4);
    float*    combB  = (float*)   carve((size_t)NL * NH * 4);
    float*    gsum   = (float*)   carve((size_t)G * HD * 4);
    float*    cnt    = (float*)   carve((size_t)G * 4);
    // pre-transposed bf16 weights
    __bf16* nWt = (__bf16*)carve((size_t)HD * NDp * 2);
    __bf16* WqT = (__bf16*)carve((size_t)NL * HD * HD * 2);
    __bf16* WkT = (__bf16*)carve((size_t)NL * HD * HD * 2);
    __bf16* WvT = (__bf16*)carve((size_t)NL * HD * HD * 2);
    __bf16* WoT = (__bf16*)carve((size_t)NL * HD * HD * 2);
    __bf16* W1T = (__bf16*)carve((size_t)NL * FFD * HD * 2);
    __bf16* W2T = (__bf16*)carve((size_t)NL * HD * FFD * 2);

    const float scale = 1.f / sqrtf((float)DH);
    const int mt = N / 16;                                // M tiles (N % 16 == 0)
    const dim3 gB(256);
    const int nhd_blocks = (N * HD + 255) / 256;
    const int eh_blocks  = (Ecnt * NH + 255) / 256;
    const int wblk  = (HD * HD + 255) / 256;
    const int wblkF = (FFD * HD + 255) / 256;

    // ----- one-time prep: pad h, transpose+cvt all weights, fused edge bias -----
    pad_rows<<<(N * NDp + 255) / 256, gB, 0, stream>>>(h, hpad, N, ND, NDp);
    transpose_cvt<<<(HD * NDp + 255) / 256, gB, 0, stream>>>(node_W, nWt, ND, NDp, HD);
    for (int l = 0; l < NL; ++l) {
        const long WW = (long)l * HD * HD, WF = (long)l * HD * FFD;
        transpose_cvt<<<wblk,  gB, 0, stream>>>(Wq + WW, WqT + WW, HD, HD, HD);
        transpose_cvt<<<wblk,  gB, 0, stream>>>(Wk + WW, WkT + WW, HD, HD, HD);
        transpose_cvt<<<wblk,  gB, 0, stream>>>(Wv + WW, WvT + WW, HD, HD, HD);
        transpose_cvt<<<wblk,  gB, 0, stream>>>(Wo + WW, WoT + WW, HD, HD, HD);
        transpose_cvt<<<wblkF, gB, 0, stream>>>(W1 + WF, W1T + WF, HD, HD, FFD);   // Wt[512][128]
        transpose_cvt<<<wblkF, gB, 0, stream>>>(W2 + WF, W2T + WF, FFD, FFD, HD);  // Wt[128][512]
    }
    build_comb<<<1, 256, 0, stream>>>(edge_W, edge_b, Web, beb, combW, combB);

    // ----- x = h @ node_W + node_b -----
    gemm_bf16_wmma<<<dim3(1, mt), 128, 0, stream>>>(hpad, nWt, node_b, nullptr, x, N, NDp, HD, 0);

    for (int l = 0; l < NL; ++l) {
        const long WW = (long)l * HD * HD, WF = (long)l * HD * FFD;
        ln_kernel<<<(N + 7) / 8, 256, 0, stream>>>(x, g1 + l * HD, be1 + l * HD, xn, N);

        gemm_bf16_wmma<<<dim3(1, mt), 128, 0, stream>>>(xn, WqT + WW, bq + l * HD, nullptr, qb, N, HD, HD, 0);
        gemm_bf16_wmma<<<dim3(1, mt), 128, 0, stream>>>(xn, WkT + WW, bk + l * HD, nullptr, kb, N, HD, HD, 0);
        gemm_bf16_wmma<<<dim3(1, mt), 128, 0, stream>>>(xn, WvT + WW, bv + l * HD, nullptr, vb, N, HD, HD, 0);

        init_attn<<<nhd_blocks, gB, 0, stream>>>(nmax, den, out_un, N);
        edge_score<<<eh_blocks, gB, 0, stream>>>(qb, kb, eg, ei, combW + l * 4 * NH,
                                                 combB + l * NH, sbuf, nmax, Ecnt, scale);
        edge_accum<<<eh_blocks, gB, 0, stream>>>(vb, ei, sbuf, nmax, den, out_un, Ecnt);
        attn_norm<<<nhd_blocks, gB, 0, stream>>>(out_un, den, N);

        // x = out @ Wo + bo + x   (res aliases C: each element read-then-written once)
        gemm_bf16_wmma<<<dim3(1, mt), 128, 0, stream>>>(out_un, WoT + WW, bo + l * HD, x, x, N, HD, HD, 0);

        ln_kernel<<<(N + 7) / 8, 256, 0, stream>>>(x, g2 + l * HD, be2 + l * HD, xn, N);
        gemm_bf16_wmma<<<dim3(4, mt), 128, 0, stream>>>(xn, W1T + WF, b1 + l * FFD, nullptr, ffn_t, N, HD, FFD, 1);
        gemm_bf16_wmma<<<dim3(1, mt), 128, 0, stream>>>(ffn_t, W2T + WF, b2 + l * HD, x, x, N, FFD, HD, 0);
    }

    // final LN straight into d_out[0 : N*HD]
    float* xout = (float*)d_out;
    ln_kernel<<<(N + 7) / 8, 256, 0, stream>>>(x, gF, bF, xout, N);

    // graph mean pooling into d_out[N*HD : ]
    fzero<<<(G * HD + 255) / 256, gB, 0, stream>>>(gsum, G * HD);
    fzero<<<(G + 255) / 256, gB, 0, stream>>>(cnt, G);
    pool_accum<<<nhd_blocks, gB, 0, stream>>>(xout, batch, gsum, cnt, N);
    pool_final<<<(G * HD + 255) / 256, gB, 0, stream>>>(gsum, cnt, xout + (long)N * HD, G);
}
